// RWA_87213605913062
// MI455X (gfx1250) — matrix operand order repaired and checked by reading, not compiled, not run on hardware
//
#include <hip/hip_runtime.h>

#define B_  32
#define T_  4096
#define F_  128
#define C_  256
#define K_  128
#define FC_ 384
#define RS_ 392   // xh LDS row stride (bf16): 384 + 8 pad  -> 4-bank row skew
#define GR_ 264   // weight LDS row stride (bf16): 256 + 8 pad -> 4-bank row skew

#define XH_ELEMS (B_ * RS_)        // 12544
#define G_ELEMS  (FC_ * GR_)       // 101376
#define U_ELEMS  (F_ * GR_)        // 33792
#define SMEM_BYTES ((XH_ELEMS + G_ELEMS + U_ELEMS) * 2)   // 295424 <= 320KB WGP LDS

typedef __attribute__((ext_vector_type(16))) __bf16 v16bf;
typedef __attribute__((ext_vector_type(8)))  __bf16 v8bf;
typedef __attribute__((ext_vector_type(8)))  float  v8f;
typedef __attribute__((ext_vector_type(4)))  float  v4f;
typedef unsigned int u32x4 __attribute__((ext_vector_type(4)));
typedef int          i32x8 __attribute__((ext_vector_type(8)));
typedef int          i32x4 __attribute__((ext_vector_type(4)));

#if defined(__has_builtin)
#if __has_builtin(__builtin_amdgcn_tensor_load_to_lds) && __has_builtin(__builtin_amdgcn_s_wait_tensorcnt)
#define USE_TDM 1
#endif
#endif
#ifndef USE_TDM
#define USE_TDM 0
#endif

static __device__ __forceinline__ v16bf cat16(v8bf lo, v8bf hi) {
  return __builtin_shufflevector(lo, hi, 0,1,2,3,4,5,6,7,8,9,10,11,12,13,14,15);
}

// Defeat LICM across the t-loop without constraining in-iteration scheduling.
static __device__ __forceinline__ const __bf16* opaque(const __bf16* p) {
  unsigned long long q = (unsigned long long)(uintptr_t)p;
  asm volatile("" : "+v"(q));
  return (const __bf16*)(uintptr_t)q;
}

// A fragment (16x32 bf16) from row-major LDS xh.
// lanes 0-15: rows M=0..15, K in [0,8)+[16,24); lanes 16-31: same rows, K in [8,16)+[24,32).
static __device__ __forceinline__ v16bf load_a(const __bf16* lds, int lane, int mbase, int k0) {
  const int row = mbase + (lane & 15);
  const int kh  = (lane >> 4) << 3;
  const __bf16* p = lds + row * RS_ + k0 + kh;
  return cat16(*(const v8bf*)(p), *(const v8bf*)(p + 16));   // 2x ds_load_b128
}

// B fragment (32x16 bf16) from K-major weights in LDS (padded rows).
static __device__ __forceinline__ v16bf load_b_lds(const __bf16* w, int lane, int k0, int nbase) {
  const __bf16* p = w + (k0 + ((lane >> 4) << 3)) * GR_ + nbase + (lane & 15);
  v16bf r;
#pragma unroll
  for (int i = 0; i < 8; ++i) { r[i] = p[i * GR_]; r[8 + i] = p[(16 + i) * GR_]; }
  return r;    // strided scalar ds loads, table k order
}

// B fragment (32x16 bf16) from K-major weights resident in L2.
static __device__ __forceinline__ v16bf load_b_g(const __bf16* wT, int lane, int k0, int ncols, int nbase) {
  const __bf16* p = wT + (size_t)(k0 + ((lane >> 4) << 3)) * ncols + nbase + (lane & 15);
  v16bf r;
#pragma unroll
  for (int i = 0; i < 8; ++i) { r[i] = p[(size_t)i * ncols]; r[8 + i] = p[(size_t)(16 + i) * ncols]; }
  return r;    // strided scalar global loads, table k order
}

static __device__ __forceinline__ v8f wmma_bf16(v16bf a, v16bf b, v8f c) {
  return __builtin_amdgcn_wmma_f32_16x16x32_bf16(false, a, false, b, (short)0, c, false, false);
}

#if USE_TDM
// TDM: DMA a 2-D bf16 image [rows x 256] from global into LDS with row padding
// (128 DWORDs data + 4 DWORDs pad per row -> GR_ = 264 bf16 stride).
static __device__ __forceinline__ void tdm_load_2d(const __bf16* gsrc, __bf16* ldst, int rows) {
  const unsigned long long ga = (unsigned long long)(uintptr_t)gsrc;
  const unsigned int       la = (unsigned int)(uintptr_t)ldst;   // LDS offset = addr[31:0]
  u32x4 g0;
  g0.x = 1u;                                                 // count=1, user descriptor
  g0.y = la;                                                 // lds_addr
  g0.z = (unsigned int)(ga & 0xFFFFFFFFu);                   // global_addr[31:0]
  g0.w = (unsigned int)((ga >> 32) & 0x01FFFFFFu) | (2u << 30);  // global_addr[56:32] | type=2
  i32x8 g1;
  g1[0] = (1 << 16) | (1 << 20) | (6 << 22) | (3 << 25);     // data_size=2B, pad_en, int=128DW, amt=4DW
  g1[1] = (int)(256u << 16);                                 // tensor_dim0=256 (low 16 -> [31:16])
  g1[2] = (int)(((unsigned)rows & 0xFFFFu) << 16);           // tensor_dim1 low16
  g1[3] = (int)(256u << 16);                                 // tile_dim0=256
  g1[4] = rows;                                              // tile_dim1=rows, tile_dim2=0
  g1[5] = 256;                                               // tensor_dim0_stride=256
  g1[6] = 0;
  g1[7] = 0;
  const i32x4 z4 = {0, 0, 0, 0};
  const i32x8 z8 = {0, 0, 0, 0, 0, 0, 0, 0};
  __builtin_amdgcn_tensor_load_to_lds(g0, g1, z4, z4, z8, 0);  // 6-arg (clang-23) form
}
#endif

// One-shot weight conversion: fp32 [N,K] row-major -> bf16 [K,N] (K-major), into d_ws.
__global__ void rwa_prep_weights(const float* __restrict__ g_w, const float* __restrict__ u_w,
                                 const float* __restrict__ a_w, const float* __restrict__ o_w,
                                 __bf16* __restrict__ gT, __bf16* __restrict__ aT,
                                 __bf16* __restrict__ uT, __bf16* __restrict__ oT) {
  const int tid = blockIdx.x * blockDim.x + threadIdx.x;
  if (tid < FC_ * C_) {
    const int k = tid / C_, n = tid % C_;
    gT[tid] = (__bf16)g_w[n * FC_ + k];
    aT[tid] = (__bf16)a_w[n * FC_ + k];
  }
  if (tid < F_ * C_) {
    const int k = tid / C_, n = tid % C_;
    uT[tid] = (__bf16)u_w[n * F_ + k];
  }
  if (tid < C_ * K_) {
    const int c = tid / K_, n = tid % K_;
    oT[tid] = (__bf16)o_w[n * C_ + c];
  }
}

// Persistent single-workgroup scan: 16 wave32 waves, one WGP.
__global__ __launch_bounds__(512, 1)
void rwa_scan_kernel(const float* __restrict__ x,
                     const float* __restrict__ n0,  const float* __restrict__ d0,
                     const float* __restrict__ h0,  const float* __restrict__ am0,
                     const float* __restrict__ g_b, const float* __restrict__ u_b,
                     const float* __restrict__ o_b,
                     const __bf16* __restrict__ gT, const __bf16* __restrict__ aT,
                     const __bf16* __restrict__ uT, const __bf16* __restrict__ oT,
                     float* __restrict__ out) {
  extern __shared__ __bf16 smem[];
  __bf16* sXH = smem;                      // [32][392]  x(0:128) | h(128:384)
  __bf16* sG  = smem + XH_ELEMS;           // [384][264] g weights, K-major
  __bf16* sU  = sG + G_ELEMS;              // [128][264] u weights, K-major

  const int tid  = threadIdx.x;
  const int lane = tid & 31;
  const int w    = tid >> 5;               // wave 0..15

  // ---- stage LDS-resident weights ----
#if USE_TDM
  if (w == 0) {
    tdm_load_2d(gT, sG, FC_);
    tdm_load_2d(uT, sU, F_);
    __builtin_amdgcn_s_wait_tensorcnt(0);
  }
#else
  for (int i = tid; i < FC_ * 32; i += 512) {
    const int r = i >> 5, c = (i & 31) << 3;
    *(v8bf*)(sG + r * GR_ + c) = *(const v8bf*)(gT + r * C_ + c);
  }
  for (int i = tid; i < F_ * 32; i += 512) {
    const int r = i >> 5, c = (i & 31) << 3;
    *(v8bf*)(sU + r * GR_ + c) = *(const v8bf*)(uT + r * C_ + c);
  }
#endif

  // ---- gate-phase assignment: wave w owns C columns [16w, 16w+16) ----
  const int   nb   = w << 4;
  const int   colC = nb + (lane & 15);
  const float gb_l = g_b[colC];
  const float ub_l = u_b[colC];

  // ---- output-phase assignment: wave w owns tile (mtO, ntO) of [32 x 128] ----
  const int   mtO  = (w >> 3) << 4;
  const int   ntO  = (w & 7) << 4;
  const float ob_l = o_b[ntO + (lane & 15)];

  // ---- per-wave persistent state registers (C/D f32 layout) ----
  const int rsub = (lane >> 4) << 3;
  float nS[2][8], dS[2][8], aM[2][8];
#pragma unroll
  for (int mt = 0; mt < 2; ++mt)
#pragma unroll
    for (int j = 0; j < 8; ++j) {
      const int b = mt * 16 + rsub + j;
      nS[mt][j] = n0 [b * C_ + colC];
      dS[mt][j] = d0 [b * C_ + colC];
      aM[mt][j] = am0[b * C_ + colC];
    }

  // ---- stage initial h into LDS ----
#pragma unroll
  for (int i = 0; i < 16; ++i) {
    const int idx = tid * 16 + i;
    const int b = idx >> 8, c = idx & 255;
    sXH[b * RS_ + F_ + c] = (__bf16)h0[b * C_ + c];
  }
  __syncthreads();

  const int xb_row = (tid * 8) >> 7;
  const int xb_col = (tid * 8) & 127;

  for (int t = 0; t < T_; ++t) {
    // Laundered bases: per-step L2 streaming of a/o weights, not hoistable/spillable.
    const __bf16* aTt = opaque(aT);
    const __bf16* oTt = opaque(oT);

    // ---- stage x_t (fp32 -> bf16), 8 elems/thread ----
    {
      const float* px = x + (size_t)xb_row * T_ * F_ + (size_t)t * F_ + xb_col;
      v4f x0 = __builtin_nontemporal_load((const v4f*)px);
      v4f x1 = __builtin_nontemporal_load((const v4f*)(px + 4));
      v8bf xb;
      xb[0] = (__bf16)x0[0]; xb[1] = (__bf16)x0[1]; xb[2] = (__bf16)x0[2]; xb[3] = (__bf16)x0[3];
      xb[4] = (__bf16)x1[0]; xb[5] = (__bf16)x1[1]; xb[6] = (__bf16)x1[2]; xb[7] = (__bf16)x1[3];
      *(v8bf*)(sXH + xb_row * RS_ + xb_col) = xb;
      if (t + 1 < T_) __builtin_prefetch(px + F_, 0, 1);
    }
    __syncthreads();  // x_t visible; all waves past previous step's phase-3 reads

    // ---- phase 1: u/g/a GEMMs over xh = [x_t | h_{t-1}] ----
    v8f aG[2] = {{}, {}}, aA[2] = {{}, {}}, aU[2] = {{}, {}};
#pragma unroll
    for (int ks = 0; ks < 12; ++ks) {
      const int k0 = ks * 32;
      v16bf af0 = load_a(sXH, lane, 0,  k0);
      v16bf af1 = load_a(sXH, lane, 16, k0);
      v16bf bg  = load_b_lds(sG, lane, k0, nb);           // LDS-resident g
      aG[0] = wmma_bf16(af0, bg, aG[0]);
      aG[1] = wmma_bf16(af1, bg, aG[1]);
      v16bf ba  = load_b_g(aTt, lane, k0, C_, nb);        // L2-streamed a
      aA[0] = wmma_bf16(af0, ba, aA[0]);
      aA[1] = wmma_bf16(af1, ba, aA[1]);
      if (ks < 4) {
        v16bf bu = load_b_lds(sU, lane, k0, nb);          // LDS-resident u
        aU[0] = wmma_bf16(af0, bu, aU[0]);
        aU[1] = wmma_bf16(af1, bu, aU[1]);
      }
    }

    // ---- phase 2: elementwise state update (registers only) ----
    float hNew[2][8];
#pragma unroll
    for (int mt = 0; mt < 2; ++mt)
#pragma unroll
      for (int j = 0; j < 8; ++j) {
        const float u  = aU[mt][j] + ub_l;
        const float g  = aG[mt][j] + gb_l;
        const float a  = aA[mt][j];
        const float z  = u * tanhf(g);
        const float an = fmaxf(aM[mt][j], a);
        const float e1 = __expf(aM[mt][j] - an);
        const float e2 = __expf(a - an);
        const float nn = nS[mt][j] * e1 + z * e2;
        const float dd = dS[mt][j] * e1 + e2;
        nS[mt][j] = nn; dS[mt][j] = dd; aM[mt][j] = an;
        hNew[mt][j] = tanhf(nn / dd);
      }

    __syncthreads();  // all phase-1 reads of h_{t-1} complete before overwrite

#pragma unroll
    for (int mt = 0; mt < 2; ++mt)
#pragma unroll
      for (int j = 0; j < 8; ++j) {
        const int r = mt * 16 + rsub + j;
        sXH[r * RS_ + F_ + colC] = (__bf16)hNew[mt][j];
      }
    __syncthreads();  // h_t visible

    // ---- phase 3: out_t = h_t @ o_w^T + o_b, one 16x16 tile per wave ----
    v8f aO = {};
#pragma unroll
    for (int ks = 0; ks < 8; ++ks) {
      const int k0 = ks * 32;
      v16bf af = load_a(sXH, lane, mtO, F_ + k0);
      v16bf bo = load_b_g(oTt, lane, k0, K_, ntO);        // L2-streamed o
      aO = wmma_bf16(af, bo, aO);
    }
#pragma unroll
    for (int j = 0; j < 8; ++j) {
      const int b = mtO + rsub + j;
      float* po = out + (size_t)b * T_ * K_ + (size_t)t * K_ + ntO + (lane & 15);
      __builtin_nontemporal_store(aO[j] + ob_l, po);
    }
  }

  // ---- final carry: (n_t, d_t, h_t, a_new), each [B, C] ----
  float* st = out + (size_t)B_ * T_ * K_;
#pragma unroll
  for (int mt = 0; mt < 2; ++mt)
#pragma unroll
    for (int j = 0; j < 8; ++j) {
      const int b = mt * 16 + rsub + j;
      const size_t o = (size_t)b * C_ + colC;
      st[o]               = nS[mt][j];
      st[B_ * C_ + o]     = dS[mt][j];
      st[2 * B_ * C_ + o] = tanhf(nS[mt][j] / dS[mt][j]);
      st[3 * B_ * C_ + o] = aM[mt][j];
    }
}

extern "C" void kernel_launch(void* const* d_in, const int* in_sizes, int n_in,
                              void* d_out, int out_size, void* d_ws, size_t ws_size,
                              hipStream_t stream) {
  const float* x   = (const float*)d_in[0];
  const float* n0  = (const float*)d_in[1];
  const float* d0  = (const float*)d_in[2];
  const float* h0  = (const float*)d_in[3];
  const float* am0 = (const float*)d_in[4];
  const float* g_w = (const float*)d_in[5];
  const float* g_b = (const float*)d_in[6];
  const float* u_w = (const float*)d_in[7];
  const float* u_b = (const float*)d_in[8];
  const float* a_w = (const float*)d_in[9];
  const float* o_w = (const float*)d_in[10];
  const float* o_b = (const float*)d_in[11];
  float* out = (float*)d_out;

  __bf16* gT = (__bf16*)d_ws;
  __bf16* aT = gT + FC_ * C_;
  __bf16* uT = aT + FC_ * C_;
  __bf16* oT = uT + F_ * C_;

  rwa_prep_weights<<<(FC_ * C_) / 256, 256, 0, stream>>>(g_w, u_w, a_w, o_w, gT, aT, uT, oT);
  rwa_scan_kernel<<<1, 512, SMEM_BYTES, stream>>>(x, n0, d0, h0, am0, g_b, u_b, o_b,
                                                  gT, aT, uT, oT, out);
}
